// BertAttention_preLN_48988396978495
// MI455X (gfx1250) — matrix operand order repaired and checked by reading, not compiled, not run on hardware
//
#include <hip/hip_runtime.h>

typedef __attribute__((ext_vector_type(16))) _Float16 v16h;
typedef __attribute__((ext_vector_type(8)))  _Float16 v8h;
typedef __attribute__((ext_vector_type(4)))  _Float16 v4h;
typedef __attribute__((ext_vector_type(8)))  float    v8f;
typedef __attribute__((ext_vector_type(4)))  unsigned int uvec4;
typedef __attribute__((ext_vector_type(4)))  int          ivec4;
typedef __attribute__((ext_vector_type(8)))  int          ivec8;

#define WMMA_F32_F16(a, b, c) \
  __builtin_amdgcn_wmma_f32_16x16x32_f16(false, (a), false, (b), (short)0, (c), false, false)

// ---------------------------------------------------------------------------
// Tensor Data Mover: 2D f16 tile (width x height elements, row stride in
// elements) global -> LDS, with LDS row padding (pad codes per D# spec:
// pad_interval code c => pad after 2^(c+1) dwords; pad_amount code a => a+1
// dwords of padding). Descriptor layout per CDNA5 ISA.
// ---------------------------------------------------------------------------
__device__ __forceinline__ void tdm_load_2d_f16(
    unsigned lds_byte_off, const void* gptr,
    unsigned width_elems, unsigned height_rows, unsigned row_stride_elems,
    unsigned pad_interval_code, unsigned pad_amount_code) {
  const unsigned long long ga = (unsigned long long)gptr;
  uvec4 g0;
  g0[0] = 1u;                                   // count=1, no gather, user desc
  g0[1] = lds_byte_off;                         // lds_addr  (bits 63:32)
  g0[2] = (unsigned)(ga & 0xFFFFFFFFull);       // global_addr[31:0]  (95:64)
  g0[3] = (unsigned)((ga >> 32) & 0x1FFFFFFull) // global_addr[56:32] (120:96)
          | (2u << 30);                         // type=2 ("image")   (127:126)
  ivec8 g1;
  g1[0] = (int)((1u << 16)                      // data_size = 2 bytes
                | (1u << 20)                    // pad_enable
                | (pad_interval_code << 22)
                | (pad_amount_code << 25));
  g1[1] = (int)((width_elems & 0xFFFFu) << 16);             // tensor_dim0 lo
  g1[2] = (int)(((width_elems >> 16) & 0xFFFFu)
                | ((height_rows & 0xFFFFu) << 16));         // dim0 hi | dim1 lo
  g1[3] = (int)(((height_rows >> 16) & 0xFFFFu)
                | ((width_elems & 0xFFFFu) << 16));         // dim1 hi | tile_dim0
  g1[4] = (int)(height_rows & 0xFFFFu);                     // tile_dim1 (tile_dim2=0)
  g1[5] = (int)row_stride_elems;                            // tensor_dim0_stride lo
  g1[6] = 0;
  g1[7] = 0;
  ivec4 z4 = {0, 0, 0, 0};
#if __clang_major__ >= 23
  ivec8 z8 = {0, 0, 0, 0, 0, 0, 0, 0};
  __builtin_amdgcn_tensor_load_to_lds(g0, g1, z4, z4, z8, 0);
#else
  __builtin_amdgcn_tensor_load_to_lds(g0, g1, z4, z4, 0);
#endif
}

// ---------------------------------------------------------------------------
// Fragment loaders (wave32), layouts per CDNA5 ISA 7.12.2.
// base must be 16B aligned; stride (halves) must be a multiple of 8.
// ---------------------------------------------------------------------------
__device__ __forceinline__ v16h frag_a(const _Float16* base, int stride) {
  const int lane = threadIdx.x & 31;
  const int r = lane & 15;
  const int kb = (lane >> 4) << 3;          // 0 or 8
  const _Float16* p = base + r * stride + kb;
  v8h lo = *(const v8h*)p;
  v8h hi = *(const v8h*)(p + 16);
  v16h f;
#pragma unroll
  for (int i = 0; i < 8; ++i) { f[i] = lo[i]; f[i + 8] = hi[i]; }
  return f;
}

__device__ __forceinline__ v16h frag_b(const _Float16* base, int stride) {
  const int lane = threadIdx.x & 31;
  const int r = lane & 15;                  // N within tile
  const int kb = (lane >> 4) << 3;          // 0 or 8
  const _Float16* p = base + r * stride + kb;
  v8h lo = *(const v8h*)p;
  v8h hi = *(const v8h*)(p + 16);
  v16h f;
#pragma unroll
  for (int i = 0; i < 8; ++i) { f[i] = lo[i]; f[i + 8] = hi[i]; }
  return f;
}

// ---------------------------------------------------------------------------
// GEMM:  C[M,N] = A[M,K] @ W[K,N] + bias[N]
// Block tile 128x128, BK=32, 256 threads = 8 waves in a 4x2 grid;
// each wave: 32 rows x 64 cols = 8 WMMA accumulators.
// Double-buffered LDS: tile s+1 streams in (TDM DMA for f16 A) while the
// WMMAs for tile s execute; one barrier per K-step.
// ---------------------------------------------------------------------------
template <typename TA, bool OUT_HALF>
__global__ __launch_bounds__(256) void gemm_wmma_kernel(
    const TA* __restrict__ A, const float* __restrict__ W,
    const float* __restrict__ bias, void* __restrict__ Cout,
    int M, int N, int K) {
  __shared__ __align__(16) _Float16 At[2][128][40];  // 32-half rows + 8 pad
  __shared__ __align__(16) _Float16 Bt[2][128][40];  // stored [n][k]

  const int m0 = blockIdx.x * 128;
  const int n0 = blockIdx.y * 128;
  const int tid = threadIdx.x;
  const int wave = tid >> 5;
  const int lane = tid & 31;
  const int wm = wave & 3;                  // 0..3 -> 32-row slice
  const int wn = wave >> 2;                 // 0..1 -> 64-col slice
  const int nl = lane & 15;
  const int mb = (lane >> 4) << 3;

  auto load_tiles = [&](int k0, int buf) {
    // --- A tile (128x32) -> f16 LDS ---
    if constexpr (sizeof(TA) == 2) {
      // f16 source: async DMA via Tensor Data Mover (rows 32 halves = 16
      // dwords -> pad_interval code 3; pad 4 dwords -> pad_amount code 3).
      if (wave == 0)
        tdm_load_2d_f16((unsigned)(size_t)&At[buf][0][0],
                        (const _Float16*)A + (size_t)m0 * K + k0,
                        32u, 128u, (unsigned)K, 3u, 3u);
    } else {
      const int rr = tid >> 3;              // 0..31
      const int c0 = (tid & 7) << 2;        // 0,4,..,28
#pragma unroll
      for (int p = 0; p < 4; ++p) {
        const int row = rr + 32 * p;
        const float4 x =
            *(const float4*)((const float*)A + (size_t)(m0 + row) * K + k0 + c0);
        v4h hx;
        hx[0] = (_Float16)x.x; hx[1] = (_Float16)x.y;
        hx[2] = (_Float16)x.z; hx[3] = (_Float16)x.w;
        *(v4h*)&At[buf][row][c0] = hx;
      }
    }
    // --- W tile (32x128), transposed -> Bt[n][k] (fp32 -> f16) ---
    {
      const int n = tid & 127;
      const int kk = tid >> 7;              // 0..1
#pragma unroll
      for (int i = 0; i < 16; ++i) {
        const int k = kk + 2 * i;           // 0..31
        Bt[buf][n][k] = (_Float16)W[(size_t)(k0 + k) * N + n0 + n];
      }
    }
  };

  v8f acc[2][4];
#pragma unroll
  for (int mi = 0; mi < 2; ++mi)
#pragma unroll
    for (int t = 0; t < 4; ++t)
#pragma unroll
      for (int e = 0; e < 8; ++e) acc[mi][t][e] = 0.0f;

  // prologue: fill buffer 0
  load_tiles(0, 0);
  if constexpr (sizeof(TA) == 2) {
    if (wave == 0) __builtin_amdgcn_s_wait_tensorcnt(0);
  }
  __syncthreads();

  const int nsteps = K >> 5;
  for (int s = 0; s < nsteps; ++s) {
    const int buf = s & 1;
    // prefetch next tile into the other buffer (overlaps the WMMAs below)
    if (s + 1 < nsteps) load_tiles((s + 1) << 5, buf ^ 1);

    const v16h af0 = frag_a(&At[buf][wm * 32][0], 40);
    const v16h af1 = frag_a(&At[buf][wm * 32 + 16][0], 40);
#pragma unroll
    for (int t = 0; t < 4; ++t) {
      const v16h bf = frag_b(&Bt[buf][wn * 64 + 16 * t][0], 40);
      acc[0][t] = WMMA_F32_F16(af0, bf, acc[0][t]);
      acc[1][t] = WMMA_F32_F16(af1, bf, acc[1][t]);
    }

    if constexpr (sizeof(TA) == 2) {
      if (s + 1 < nsteps && wave == 0) __builtin_amdgcn_s_wait_tensorcnt(0);
    }
    __syncthreads();   // publishes buf^1, retires all reads of buf
  }

  // --- epilogue: bias add + store ---
#pragma unroll
  for (int mi = 0; mi < 2; ++mi)
#pragma unroll
    for (int t = 0; t < 4; ++t) {
      const int gn = n0 + wn * 64 + 16 * t + nl;
      const float bv = bias[gn];
#pragma unroll
      for (int e = 0; e < 8; ++e) {
        const int gm = m0 + wm * 32 + mi * 16 + mb + e;
        const float v = acc[mi][t][e] + bv;
        if constexpr (OUT_HALF)
          ((_Float16*)Cout)[(size_t)gm * N + gn] = (_Float16)v;
        else
          ((float*)Cout)[(size_t)gm * N + gn] = v;
      }
    }
}

// ---------------------------------------------------------------------------
// Flash attention: one block = (b, h, 128 query rows); 8 waves x 16 rows.
// Double-buffered K/V tiles: TDM DMA + V-transpose for key-tile s+1 stream in
// while the 16 WMMAs + online softmax for tile s execute.
// ---------------------------------------------------------------------------
__global__ __launch_bounds__(256) void attn_wmma_kernel(
    const _Float16* __restrict__ Qg, const _Float16* __restrict__ Kg,
    const _Float16* __restrict__ Vg, const float* __restrict__ mask,
    _Float16* __restrict__ Ctx) {
  constexpr int L = 2048, D = 1024, HD = 64;
  __shared__ __align__(16) _Float16 Kt[2][64][72];    // [key][hd] (TDM dest)
  __shared__ __align__(16) _Float16 Vt[2][64][72];    // [hd][key] (transposed)
  __shared__ __align__(16) _Float16 Pl[8][16][72];    // per-wave probs [m][key]

  const int qt = blockIdx.x, h = blockIdx.y, b = blockIdx.z;
  const int tid = threadIdx.x;
  const int wave = tid >> 5;
  const int lane = tid & 31;
  const int nl = lane & 15;
  const int lh = lane >> 4;
  const int q0 = qt * 128;

  auto load_kv = [&](int j0, int buf) {
    // K tile [64 keys][64 hd] via Tensor Data Mover (async DMA).
    // rows 64 halves = 32 dwords -> pad_interval code 4; pad 4 dwords -> 3.
    if (wave == 0)
      tdm_load_2d_f16((unsigned)(size_t)&Kt[buf][0][0],
                      Kg + (size_t)(b * L + j0) * D + h * HD,
                      64u, 64u, (unsigned)D, 4u, 3u);
    // V tile transposed -> Vt[hd][key] (overlaps the DMA).
#pragma unroll
    for (int i = 0; i < 16; ++i) {
      const int idx = i * 256 + tid;
      const int hd = idx & 63, kr = idx >> 6;
      Vt[buf][hd][kr] = Vg[(size_t)(b * L + j0 + kr) * D + h * HD + hd];
    }
  };

  // Q fragments for this wave's 16 rows, full HD=64 (2 k-steps), from global.
  v16h qf[2];
  {
    const _Float16* qbase =
        Qg + (size_t)(b * L + q0 + wave * 16 + nl) * D + h * HD;
#pragma unroll
    for (int ks = 0; ks < 2; ++ks) {
      const int kb = lh * 8 + ks * 32;
      const v8h lo = *(const v8h*)(qbase + kb);
      const v8h hi = *(const v8h*)(qbase + kb + 16);
#pragma unroll
      for (int i = 0; i < 8; ++i) { qf[ks][i] = lo[i]; qf[ks][i + 8] = hi[i]; }
    }
  }

  float mrow[8], lrow[8];
  v8f O[4];
#pragma unroll
  for (int e = 0; e < 8; ++e) { mrow[e] = -1e30f; lrow[e] = 0.0f; }
#pragma unroll
  for (int t = 0; t < 4; ++t)
#pragma unroll
    for (int e = 0; e < 8; ++e) O[t][e] = 0.0f;

  // prologue: fill buffer 0 with key-tile 0
  load_kv(0, 0);
  if (wave == 0) __builtin_amdgcn_s_wait_tensorcnt(0);
  __syncthreads();

  constexpr int NT = L / 64;   // 32 key tiles
  for (int s = 0; s < NT; ++s) {
    const int j0 = s * 64;
    const int buf = s & 1;
    // prefetch next K/V tile (DMA + transpose) into the other buffer
    if (s + 1 < NT) load_kv(j0 + 64, buf ^ 1);

    // --- S = (Q K^T) * scale + mask ---
    v8f S[4];
#pragma unroll
    for (int t = 0; t < 4; ++t) {
      const v16h b0 = frag_b(&Kt[buf][16 * t][0], 72);
      const v16h b1 = frag_b(&Kt[buf][16 * t][0] + 32, 72);
      v8f z;
#pragma unroll
      for (int e = 0; e < 8; ++e) z[e] = 0.0f;
      S[t] = WMMA_F32_F16(qf[0], b0, z);
      S[t] = WMMA_F32_F16(qf[1], b1, S[t]);
      const float mval = mask[b * L + j0 + 16 * t + nl];
#pragma unroll
      for (int e = 0; e < 8; ++e) S[t][e] = S[t][e] * 0.125f + mval;
    }

    // --- online softmax (row = reg index e within 16-lane halves) ---
    float mnew[8], alpha[8], rsum[8];
#pragma unroll
    for (int e = 0; e < 8; ++e) {
      float v = fmaxf(fmaxf(S[0][e], S[1][e]), fmaxf(S[2][e], S[3][e]));
      v = fmaxf(v, __shfl_xor(v, 1));
      v = fmaxf(v, __shfl_xor(v, 2));
      v = fmaxf(v, __shfl_xor(v, 4));
      v = fmaxf(v, __shfl_xor(v, 8));
      mnew[e] = fmaxf(mrow[e], v);
      alpha[e] = __expf(mrow[e] - mnew[e]);
      rsum[e] = 0.0f;
    }
#pragma unroll
    for (int t = 0; t < 4; ++t)
#pragma unroll
      for (int e = 0; e < 8; ++e) {
        const float p = __expf(S[t][e] - mnew[e]);
        rsum[e] += p;
        Pl[wave][e + 8 * lh][16 * t + nl] = (_Float16)p;
      }
#pragma unroll
    for (int e = 0; e < 8; ++e) {
      float s2 = rsum[e];
      s2 += __shfl_xor(s2, 1);
      s2 += __shfl_xor(s2, 2);
      s2 += __shfl_xor(s2, 4);
      s2 += __shfl_xor(s2, 8);
      lrow[e] = lrow[e] * alpha[e] + s2;
      mrow[e] = mnew[e];
    }
#pragma unroll
    for (int t = 0; t < 4; ++t)
#pragma unroll
      for (int e = 0; e < 8; ++e) O[t][e] *= alpha[e];

    // --- O += P @ V ---
#pragma unroll
    for (int ks = 0; ks < 2; ++ks) {
      const v16h af = frag_a(&Pl[wave][0][0] + ks * 32, 72);
#pragma unroll
      for (int t = 0; t < 4; ++t) {
        const v16h vf = frag_b(&Vt[buf][16 * t][0] + ks * 32, 72);
        O[t] = WMMA_F32_F16(af, vf, O[t]);
      }
    }

    if (s + 1 < NT && wave == 0) __builtin_amdgcn_s_wait_tensorcnt(0);
    __syncthreads();   // publishes buf^1, retires all reads of buf
  }

  // --- epilogue: normalize, store context (f16) ---
#pragma unroll
  for (int t = 0; t < 4; ++t)
#pragma unroll
    for (int e = 0; e < 8; ++e) {
      const float o = O[t][e] / lrow[e];
      const size_t row = (size_t)(b * L + q0 + wave * 16 + e + 8 * lh);
      Ctx[row * D + h * HD + 16 * t + nl] = (_Float16)o;
    }
}

// ---------------------------------------------------------------------------
extern "C" void kernel_launch(void* const* d_in, const int* in_sizes, int n_in,
                              void* d_out, int out_size, void* d_ws, size_t ws_size,
                              hipStream_t stream) {
  (void)in_sizes; (void)n_in; (void)out_size; (void)ws_size;
  const float* X    = (const float*)d_in[0];
  const float* mask = (const float*)d_in[1];
  const float* Wq   = (const float*)d_in[2];
  const float* bq   = (const float*)d_in[3];
  const float* Wk   = (const float*)d_in[4];
  const float* bk   = (const float*)d_in[5];
  const float* Wv   = (const float*)d_in[6];
  const float* bv   = (const float*)d_in[7];
  const float* Wo   = (const float*)d_in[8];
  const float* bo   = (const float*)d_in[9];

  constexpr int M = 4096, N = 1024, K = 1024;   // M = B*L
  _Float16* Q = (_Float16*)d_ws;
  _Float16* Kb = Q + (size_t)M * N;
  _Float16* V = Kb + (size_t)M * N;
  _Float16* C = V + (size_t)M * N;

  const dim3 gemm_grid(M / 128, N / 128, 1);
  const dim3 blk(256, 1, 1);

  gemm_wmma_kernel<float, true><<<gemm_grid, blk, 0, stream>>>(X, Wq, bq, Q, M, N, K);
  gemm_wmma_kernel<float, true><<<gemm_grid, blk, 0, stream>>>(X, Wk, bk, Kb, M, N, K);
  gemm_wmma_kernel<float, true><<<gemm_grid, blk, 0, stream>>>(X, Wv, bv, V, M, N, K);

  attn_wmma_kernel<<<dim3(2048 / 128, 16, 2), blk, 0, stream>>>(Q, Kb, V, mask, C);

  gemm_wmma_kernel<_Float16, false><<<gemm_grid, blk, 0, stream>>>(C, Wo, bo, d_out, M, N, K);
}
